// AdvancedScaledDotProductAttention_36275293782421
// MI455X (gfx1250) — matrix operand order repaired and checked by reading, not compiled, not run on hardware
//
#include <hip/hip_runtime.h>

typedef __attribute__((ext_vector_type(16))) _Float16 v16h;
typedef __attribute__((ext_vector_type(8)))  float    v8f;

#define TQ    16      // query rows per workgroup
#define TKB   64      // keys/values staged per block
#define HD    64      // head dim
#define SEQ   2048    // sequence length
#define NTHR  128     // 4 wave32

__global__ __launch_bounds__(NTHR)
void rope_attn_kernel(const float* __restrict__ Q,
                      const float* __restrict__ K,
                      const float* __restrict__ V,
                      float* __restrict__ Out,
                      float* __restrict__ Attn)
{
    extern __shared__ char smem[];
    float*    sS   = (float*)smem;                                   // [16][2048] scores (fp32)
    _Float16* sQ   = (_Float16*)(smem + TQ * SEQ * 4);               // [16][64]  f16 RoPE'd Q * scale
    _Float16* sKV  = (_Float16*)(smem + TQ * SEQ * 4 + TQ * HD * 2); // [64][64]  f16 staging (K rows / V^T)
    float*    sRed = (float*)(smem + TQ * SEQ * 4 + TQ * HD * 2 + TKB * HD * 2); // [128]
    float*    sInv = sRed + NTHR;                                    // [16] per-row 1/sum

    const int tid  = threadIdx.x;
    const int lane = tid & 31;
    const int wave = tid >> 5;
    const int bh   = blockIdx.x >> 7;          // SEQ/TQ = 128 tiles per (b,h)
    const int tile = blockIdx.x & 127;
    const int t0   = tile * TQ;

    const float* qb = Q + (size_t)bh * SEQ * HD;
    const float* kb = K + (size_t)bh * SEQ * HD;
    const float* vb = V + (size_t)bh * SEQ * HD;
    float* outb  = Out  + (size_t)bh * SEQ * HD + (size_t)t0 * HD;
    float* attnb = Attn + (size_t)bh * SEQ * SEQ + (size_t)t0 * SEQ;

    const float LOG1E4_32 = 9.2103403719761836f / 32.0f;  // ln(10000)/32
    const float QSCALE    = 0.125f;                       // 1/sqrt(64), TEMPERATURE=1

    // Rotary index j = (p & 31) is loop-invariant per thread (stride 128 == 0 mod 32):
    // compute inv_freq once per thread instead of ~500 __expf's.
    const int   jrot = tid & 31;
    const float invf = __expf(-(float)jrot * LOG1E4_32);

    // ---- stage Q tile: RoPE + pre-scale, fp32 -> f16 (512 rotary pairs) ----
    for (int p = tid; p < TQ * 32; p += NTHR) {
        int r = p >> 5;
        float x1 = qb[(t0 + r) * HD + jrot];
        float x2 = qb[(t0 + r) * HD + jrot + 32];
        float ang = (float)(t0 + r) * invf;
        float sn, cs;
        __sincosf(ang, &sn, &cs);
        sQ[r * HD + jrot]      = (_Float16)((x1 * cs - x2 * sn) * QSCALE);
        sQ[r * HD + jrot + 32] = (_Float16)((x1 * sn + x2 * cs) * QSCALE);
    }
    __syncthreads();

    // ---- pass 1: scores strip = (Q/sqrt(D)) @ RoPE(K)^T via v_wmma_f32_16x16x32_f16 ----
    for (int blk = 0; blk < SEQ / TKB; ++blk) {
        const int s0 = blk * TKB;

        // prefetch next 64-key block (16KB = 128 threads x 1 cacheline) while
        // this block is staged/multiplied: global_prefetch_b8
        if (s0 + TKB < SEQ)
            __builtin_prefetch(kb + (size_t)(s0 + TKB) * HD + tid * 32, 0, 3);

        // stage 64 keys with RoPE (2048 rotary pairs, 16/thread)
        for (int p = tid; p < TKB * 32; p += NTHR) {
            int s = p >> 5;
            float x1 = kb[(s0 + s) * HD + jrot];
            float x2 = kb[(s0 + s) * HD + jrot + 32];
            float ang = (float)(s0 + s) * invf;
            float sn, cs;
            __sincosf(ang, &sn, &cs);
            sKV[s * HD + jrot]      = (_Float16)(x1 * cs - x2 * sn);
            sKV[s * HD + jrot + 32] = (_Float16)(x1 * sn + x2 * cs);
        }
        __syncthreads();

        // each wave owns keys [s0 + 16*wave, +16): one 16x16 C tile, K-dim = 64 (2 WMMAs)
        {
            const int mn = lane & 15;
            const int hi = lane >> 4;
            v8f c = {};
            #pragma unroll
            for (int kc = 0; kc < 2; ++kc) {
                const int k0 = kc * 32;
                v16h a, b;
                // A layout: lanes 0-15 -> K k0+{0..7,16..23}; lanes 16-31 -> K k0+{8..15,24..31}
                const _Float16* arow = sQ + mn * HD + k0 + hi * 8;
                #pragma unroll
                for (int i = 0; i < 8; ++i) a[i]     = arow[i];
                #pragma unroll
                for (int i = 0; i < 8; ++i) a[8 + i] = arow[16 + i];
                // B = K^T: lane n holds K-row (s0+16w+n); lanes 0-15 -> K k0+{0..7,16..23}, lanes 16-31 -> k0+{8..15,24..31}
                const _Float16* brow = sKV + (wave * 16 + mn) * HD + k0 + hi * 8;
                #pragma unroll
                for (int i = 0; i < 8; ++i) b[i]     = brow[i];
                #pragma unroll
                for (int i = 0; i < 8; ++i) b[8 + i] = brow[16 + i];
                c = __builtin_amdgcn_wmma_f32_16x16x32_f16(false, a, false, b,
                                                           (short)0, c, false, false);
            }
            // C layout: VGPR i -> M = i + 8*hi, N = lane&15
            #pragma unroll
            for (int i = 0; i < 8; ++i)
                sS[(i + 8 * hi) * SEQ + s0 + wave * 16 + mn] = c[i];
        }
        __syncthreads();
    }

    // ---- softmax over the 16 x 2048 strip (8 threads per row) ----
    const int row = tid >> 3;
    const int sub = tid & 7;
    float mx = -3.4e38f;
    for (int c0 = sub * 256; c0 < sub * 256 + 256; ++c0)
        mx = fmaxf(mx, sS[row * SEQ + c0]);
    sRed[tid] = mx;
    __syncthreads();
    float rmax = -3.4e38f;
    #pragma unroll
    for (int i = 0; i < 8; ++i) rmax = fmaxf(rmax, sRed[row * 8 + i]);
    __syncthreads();
    float sum = 0.f;
    for (int c0 = sub * 256; c0 < sub * 256 + 256; ++c0) {
        float e = __expf(sS[row * SEQ + c0] - rmax);
        sS[row * SEQ + c0] = e;                 // keep un-normalized exp in LDS
        sum += e;
    }
    sRed[tid] = sum;
    __syncthreads();
    float rsum = 0.f;
    #pragma unroll
    for (int i = 0; i < 8; ++i) rsum += sRed[row * 8 + i];
    if (sub == 0) sInv[row] = 1.0f / rsum;
    __syncthreads();

    // ---- stream normalized attention tile to HBM, coalesced float4 ----
    {
        const float4* src = (const float4*)sS;
        float4* dst = (float4*)attnb;
        for (int i = tid; i < TQ * SEQ / 4; i += NTHR) {
            int r = i >> 9;                     // (4*i)/2048
            float4 pv = src[i];
            float iv = sInv[r];
            pv.x *= iv; pv.y *= iv; pv.z *= iv; pv.w *= iv;
            dst[i] = pv;
        }
    }

    // ---- pass 2: Out tile = softmax(P) @ V ; wave w owns d-cols [16w, 16w+16) ----
    {
        const int mn = lane & 15;
        const int hi = lane >> 4;
        v8f acc = {};
        for (int blk = 0; blk < SEQ / TKB; ++blk) {
            const int s0 = blk * TKB;
            __syncthreads();                     // protect sKV vs previous iter reads

            // prefetch next 64x64 V block while this one is staged/multiplied
            if (s0 + TKB < SEQ)
                __builtin_prefetch(vb + (size_t)(s0 + TKB) * HD + tid * 32, 0, 3);

            // stage V transposed: sKV[d][s] = V[s0+s][d] as f16
            for (int p = tid; p < TKB * HD; p += NTHR) {
                int s = p >> 6;
                int d = p & 63;
                sKV[d * TKB + s] = (_Float16)vb[(s0 + s) * HD + d];
            }
            __syncthreads();
            #pragma unroll
            for (int sc = 0; sc < 2; ++sc) {     // K-dim = 64 keys, two 32-wide chunks
                const int sk = sc * 32;
                v16h a, b;
                // A = exp(P) rows (f32 in LDS -> f16)
                const float* prow = sS + mn * SEQ + s0 + sk + hi * 8;
                #pragma unroll
                for (int i = 0; i < 8; ++i) a[i]     = (_Float16)prow[i];
                #pragma unroll
                for (int i = 0; i < 8; ++i) a[8 + i] = (_Float16)prow[16 + i];
                // B = V: lane n holds V column d = 16w+n (row of V^T staging)
                const _Float16* brow = sKV + (wave * 16 + mn) * TKB + sk + hi * 8;
                #pragma unroll
                for (int i = 0; i < 8; ++i) b[i]     = brow[i];
                #pragma unroll
                for (int i = 0; i < 8; ++i) b[8 + i] = brow[16 + i];
                acc = __builtin_amdgcn_wmma_f32_16x16x32_f16(false, a, false, b,
                                                             (short)0, acc, false, false);
            }
        }
        // normalize by 1/rowsum at the store: C layout M = i + 8*hi, N = lane&15
        #pragma unroll
        for (int i = 0; i < 8; ++i) {
            int r = i + 8 * hi;
            outb[r * HD + wave * 16 + mn] = acc[i] * sInv[r];
        }
    }
}

extern "C" void kernel_launch(void* const* d_in, const int* in_sizes, int n_in,
                              void* d_out, int out_size, void* d_ws, size_t ws_size,
                              hipStream_t stream) {
    (void)in_sizes; (void)n_in; (void)d_ws; (void)ws_size; (void)out_size;
    const float* q = (const float*)d_in[0];
    const float* k = (const float*)d_in[1];
    const float* v = (const float*)d_in[2];
    float* out  = (float*)d_out;
    float* attn = out + (size_t)2 * 16 * SEQ * HD;   // attn follows out in d_out

    const size_t smem = (size_t)TQ * SEQ * 4      // score strip (128 KB)
                      + (size_t)TQ * HD * 2       // Q f16
                      + (size_t)TKB * HD * 2      // K/V^T f16 staging
                      + (size_t)(NTHR + TQ) * 4;  // reductions
    // Allow >64KB dynamic LDS (CDNA5 WGP has 320KB)
    hipFuncSetAttribute((const void*)rope_attn_kernel,
                        hipFuncAttributeMaxDynamicSharedMemorySize, (int)smem);

    dim3 grid(2 * 16 * (SEQ / TQ));   // 4096 workgroups: one 16-row tile each
    rope_attn_kernel<<<grid, NTHR, smem, stream>>>(q, k, v, out, attn);
}
